// KANLayer_56375740727466
// MI455X (gfx1250) — matrix operand order repaired and checked by reading, not compiled, not run on hardware
//
#include <hip/hip_runtime.h>

// ---------------------------------------------------------------------------
// Fused KAN layer for MI455X (gfx1250, wave32):
//   out[b,o] = sum_k A[b,k] * W[o,k],  K = 512 (base) + 4096 (spline basis)
// A built on the fly in LDS (k<512 -> x, else cardinal cubic B-spline basis).
// 128x128 block tile, 8 wave32s, each wave 32x64 (2x4 WMMA tiles),
// double-buffered LDS, v_wmma_f32_16x16x32_f16 with fp32 accumulation.
// K-loop split into branch-free base/spline loops with peeled boundary.
// ---------------------------------------------------------------------------

typedef __attribute__((ext_vector_type(16))) _Float16 v16h;
typedef __attribute__((ext_vector_type(8)))  float    v8f;

#define BM 128
#define BN 128
#define BK 64
#define KBASE 512
#define KTOT  4608          // 512 + 512*8
#define LDSP  72            // LDS row stride in halves (144B, 16B aligned)

// Uniform cubic B-spline (h=0.4, knots t_m=-2.2+0.4m): for x in cell
// c = floor((x+1)/h) (0..4), only basis m=c..c+3 are nonzero, with the
// cardinal weights below. Matches the reference's Cox-de Boor recursion.
__device__ __forceinline__ void bspline_pack(float x, _Float16 h8[8]) {
  float u  = (x + 1.0f) * 2.5f;
  float cf = floorf(u);
  cf = fminf(fmaxf(cf, 0.0f), 4.0f);
  const int c = (int)cf;
  const float s  = u - cf;
  const float s2 = s * s;
  const float s3 = s2 * s;
  const float om = 1.0f - s;
  const float k6 = 1.0f / 6.0f;
  const float w0 = om * om * om * k6;
  const float w1 = (3.0f * s3 - 6.0f * s2 + 4.0f) * k6;
  const float w2 = (-3.0f * s3 + 3.0f * s2 + 3.0f * s + 1.0f) * k6;
  const float w3 = s3 * k6;
#pragma unroll
  for (int m = 0; m < 8; ++m) {
    const int d = m - c;
    float v = (d == 0) ? w0 : (d == 1) ? w1 : (d == 2) ? w2 : (d == 3) ? w3
                                                                       : 0.0f;
    h8[m] = (_Float16)v;
  }
}

// A fragment (16x32 f16): lane holds row M=lane&15; K pairs per ISA table:
// u32 index in row = (p&3) + 4*half + 8*(p>>2), p = 0..7.
__device__ __forceinline__ v16h load_a_frag(const _Float16* rowp, int half) {
  union { v16h v; unsigned u[8]; } f;
  const unsigned* r = reinterpret_cast<const unsigned*>(rowp);
  const int o = 4 * half;
#pragma unroll
  for (int p = 0; p < 8; ++p)
    f.u[p] = r[(p & 3) + o + 8 * (p >> 2)];
  return f.v;
}

// B fragment (32x16 f16): lane holds col N=lane&15; K 8*half..+7 and
// 16+8*half..+7 (ISA table order, matches A) -> two 16B LDS loads.
__device__ __forceinline__ v16h load_b_frag(const _Float16* rowp, int half) {
  union { v16h v; uint4 q[2]; } f;
  const uint4* r = reinterpret_cast<const uint4*>(rowp);
  f.q[0] = r[half];
  f.q[1] = r[2 + half];
  return f.v;
}

// f32 -> f16 row stager: 64 contiguous floats per row, 2 threads per row.
__device__ __forceinline__ void stage_row32(const float* __restrict__ src,
                                            _Float16* dst) {
#pragma unroll
  for (int j = 0; j < 8; ++j) {
    float4 v = reinterpret_cast<const float4*>(src)[j];
    union { _Float16 h[4]; uint2 u; } pk;
    pk.h[0] = (_Float16)v.x; pk.h[1] = (_Float16)v.y;
    pk.h[2] = (_Float16)v.z; pk.h[3] = (_Float16)v.w;
    *reinterpret_cast<uint2*>(dst + 4 * j) = pk.u;
  }
}

// Branch-free base-region tile: A from x, B from base_weight.
__device__ __forceinline__ void load_base_tile(const float* __restrict__ x,
                                               const float* __restrict__ bw,
                                               int row0, int col0, int tid,
                                               int kc, _Float16 (*As)[LDSP],
                                               _Float16 (*Bs)[LDSP]) {
  const int lr = tid >> 1;            // 0..127
  const int lk = (tid & 1) * 32;      // {0,32}
  stage_row32(x  + (size_t)(row0 + lr) * 512 + (kc + lk), &As[lr][lk]);
  stage_row32(bw + (size_t)(col0 + lr) * 512 + (kc + lk), &Bs[lr][lk]);
}

// Branch-free spline-region tile: A = on-the-fly basis, B from spline_weight
// (spline_weight[o,i,l] is already contiguous in combined-k per o row).
__device__ __forceinline__ void load_spline_tile(const float* __restrict__ x,
                                                 const float* __restrict__ sw,
                                                 int row0, int col0, int tid,
                                                 int kc, _Float16 (*As)[LDSP],
                                                 _Float16 (*Bs)[LDSP]) {
  const int i0 = (kc - KBASE) >> 3;        // first input feature of tile
#pragma unroll
  for (int j = 0; j < 4; ++j) {
    const int idx = tid + 256 * j;         // 0..1023 = 128 rows x 8 feats
    const int r   = idx >> 3;
    const int fi  = idx & 7;
    const float xv = x[(size_t)(row0 + r) * 512 + (i0 + fi)];
    union { _Float16 h[8]; uint4 q; } pk;
    bspline_pack(xv, pk.h);
    *reinterpret_cast<uint4*>(&As[r][8 * fi]) = pk.q;     // 16B aligned
  }
  const int lr = tid >> 1;
  const int lk = (tid & 1) * 32;
  stage_row32(sw + (size_t)(col0 + lr) * 4096 + (kc - KBASE + lk),
              &Bs[lr][lk]);
}

// One BK=64 stage: 2 A-frags x 4 B-frags x 2 k-steps -> 16 v_wmma.
__device__ __forceinline__ void mma_tile(const _Float16 (*As)[LDSP],
                                         const _Float16 (*Bs)[LDSP],
                                         int wm, int wn, int l16, int half,
                                         v8f acc[2][4]) {
#pragma unroll
  for (int ks = 0; ks < BK; ks += 32) {
    v16h a0 = load_a_frag(&As[wm + l16][ks],      half);
    v16h a1 = load_a_frag(&As[wm + 16 + l16][ks], half);
    v16h b0 = load_b_frag(&Bs[wn + l16][ks],      half);
    v16h b1 = load_b_frag(&Bs[wn + 16 + l16][ks], half);
    v16h b2 = load_b_frag(&Bs[wn + 32 + l16][ks], half);
    v16h b3 = load_b_frag(&Bs[wn + 48 + l16][ks], half);
    acc[0][0] = __builtin_amdgcn_wmma_f32_16x16x32_f16(false, a0, false, b0,
                                              (short)0, acc[0][0], false, false);
    acc[0][1] = __builtin_amdgcn_wmma_f32_16x16x32_f16(false, a0, false, b1,
                                              (short)0, acc[0][1], false, false);
    acc[0][2] = __builtin_amdgcn_wmma_f32_16x16x32_f16(false, a0, false, b2,
                                              (short)0, acc[0][2], false, false);
    acc[0][3] = __builtin_amdgcn_wmma_f32_16x16x32_f16(false, a0, false, b3,
                                              (short)0, acc[0][3], false, false);
    acc[1][0] = __builtin_amdgcn_wmma_f32_16x16x32_f16(false, a1, false, b0,
                                              (short)0, acc[1][0], false, false);
    acc[1][1] = __builtin_amdgcn_wmma_f32_16x16x32_f16(false, a1, false, b1,
                                              (short)0, acc[1][1], false, false);
    acc[1][2] = __builtin_amdgcn_wmma_f32_16x16x32_f16(false, a1, false, b2,
                                              (short)0, acc[1][2], false, false);
    acc[1][3] = __builtin_amdgcn_wmma_f32_16x16x32_f16(false, a1, false, b3,
                                              (short)0, acc[1][3], false, false);
  }
}

__global__ __launch_bounds__(256)
void kan_fused_wmma(const float* __restrict__ x,
                    const float* __restrict__ bw,
                    const float* __restrict__ sw,
                    float* __restrict__ out) {
  __shared__ alignas(16) _Float16 As[2][BM][LDSP];   // 2 x 18KB
  __shared__ alignas(16) _Float16 Bs[2][BN][LDSP];   // 2 x 18KB

  const int tid  = threadIdx.x;
  const int lane = tid & 31;
  const int wave = tid >> 5;           // 0..7
  const int l16  = lane & 15;
  const int half = lane >> 4;
  const int wm   = (wave >> 1) * 32;   // row quadrant: 0,32,64,96
  const int wn   = (wave & 1)  * 64;   // col half: 0,64
  const int row0 = blockIdx.y * BM;
  const int col0 = blockIdx.x * BN;

  v8f acc[2][4] = {};                  // 2 row-tiles x 4 col-tiles

  load_base_tile(x, bw, row0, col0, tid, 0, As[0], Bs[0]);

  int p = 0;
  // ---- base region: 7 iterations, next tile is also base --------------
  for (int kc = 0; kc < KBASE - BK; kc += BK) {
    __syncthreads();
    load_base_tile(x, bw, row0, col0, tid, kc + BK, As[p ^ 1], Bs[p ^ 1]);
    mma_tile(As[p], Bs[p], wm, wn, l16, half, acc);
    p ^= 1;
  }
  // ---- boundary: compute last base tile, stage first spline tile ------
  __syncthreads();
  load_spline_tile(x, sw, row0, col0, tid, KBASE, As[p ^ 1], Bs[p ^ 1]);
  mma_tile(As[p], Bs[p], wm, wn, l16, half, acc);
  p ^= 1;
  // ---- spline region: 63 iterations, branch-free body -----------------
  for (int kc = KBASE; kc < KTOT - BK; kc += BK) {
    __syncthreads();
    load_spline_tile(x, sw, row0, col0, tid, kc + BK, As[p ^ 1], Bs[p ^ 1]);
    mma_tile(As[p], Bs[p], wm, wn, l16, half, acc);
    p ^= 1;
  }
  // ---- final tile ------------------------------------------------------
  __syncthreads();
  mma_tile(As[p], Bs[p], wm, wn, l16, half, acc);

  // -------- store: D layout M = vgpr + 8*half, N = lane&15 --------------
#pragma unroll
  for (int i = 0; i < 2; ++i) {
#pragma unroll
    for (int t = 0; t < 4; ++t) {
      const int N = col0 + wn + t * 16 + l16;
#pragma unroll
      for (int r = 0; r < 8; ++r) {
        const int M = row0 + wm + i * 16 + r + 8 * half;
        out[(size_t)M * 512 + N] = acc[i][t][r];
      }
    }
  }
}

extern "C" void kernel_launch(void* const* d_in, const int* in_sizes, int n_in,
                              void* d_out, int out_size, void* d_ws, size_t ws_size,
                              hipStream_t stream) {
  (void)in_sizes; (void)n_in; (void)d_ws; (void)ws_size; (void)out_size;
  const float* x  = (const float*)d_in[0];   // (8192, 512)
  const float* bw = (const float*)d_in[1];   // (512, 512)
  const float* sw = (const float*)d_in[2];   // (512, 512, 8)
  float* out = (float*)d_out;                // (8192, 512)
  dim3 grid(512 / BN, 8192 / BM);            // (4, 64)
  dim3 block(256);                           // 8 wave32s
  hipLaunchKernelGGL(kan_fused_wmma, grid, block, 0, stream, x, bw, sw, out);
}
